// SparseMHADecoder_40501541601693
// MI455X (gfx1250) — matrix operand order repaired and checked by reading, not compiled, not run on hardware
//
#include <hip/hip_runtime.h>
#include <hip/hip_bf16.h>
#include <math.h>

typedef __bf16 bf16;
typedef __attribute__((ext_vector_type(16))) __bf16 v16bf;
typedef __attribute__((ext_vector_type(8)))  float  v8f;
typedef __attribute__((ext_vector_type(4)))  float  f32x4;
typedef __attribute__((ext_vector_type(4)))  __bf16 bf16x4;

#define BATCH  2
#define LQ     4096
#define LKV    1024
#define DMODEL 512
#define HEADS  8
#define DHEAD  64

// ---------------------------------------------------------------------------
// GEMM: C[M,512] (f32) = A[M,512] (f32 or bf16) x W[512,512] (f32 -> bf16)
// Block tile 128x128, K-step 32, 256 threads = 8 waves.
// Wave (wm in 0..3, wn in 0..1) computes a 32x64 slab: 2x4 v_wmma tiles.
// LDS tiles are stored pre-swizzled in the CDNA5 WMMA fragment layout so each
// lane's fragment is one contiguous 32B chunk (two ds_load_b128).
//   A frag (16x32 bf16): lane m  (0..15)  -> K{0..7, 16..23} of row m
//                        lane m+16        -> K{8..15,24..31} of row m
//   B frag (32x16 bf16): lane n  (0..15)  -> K{0..7, 16..23} of col n
//                        lane n+16        -> K{8..15,24..31} of col n
// C frag (16x16 f32): VGPR p, lanes 0..15 -> (M=p,  N=lane)
//                             lanes 16..31-> (M=p+8,N=lane-16)
// ---------------------------------------------------------------------------
template <bool ABF16>
__global__ __launch_bounds__(256)
void gemm512_bf16_wmma(const void* __restrict__ A_,
                       const float* __restrict__ W,
                       float* __restrict__ C)
{
  __shared__ __align__(16) bf16 As[128 * 32];   // 8 KB, fragment order
  __shared__ __align__(16) bf16 Bs[32 * 128];   // 8 KB, fragment order

  const int tid  = threadIdx.x;
  const int lane = tid & 31;
  const int wave = tid >> 5;
  const int wm   = wave >> 1;            // 0..3 -> 32-row slab
  const int wn   = wave & 1;             // 0..1 -> 64-col slab
  const int blockN = blockIdx.x * 128;
  const int blockM = blockIdx.y * 128;

  v8f acc[2][4];
  const v8f vzero = {0.f, 0.f, 0.f, 0.f, 0.f, 0.f, 0.f, 0.f};
#pragma unroll
  for (int i = 0; i < 2; ++i)
#pragma unroll
    for (int j = 0; j < 4; ++j) acc[i][j] = vzero;

  // A staging: 4 passes of (32 rows x 8 k-vec4)
  const int a_kvec = tid & 7;            // k0 = a_kvec*4
  const int a_rowo = tid >> 3;           // 0..31
  // B staging: thread t -> column n = t&127, k half = t>>7 (16 K values)
  const int b_n    = tid & 127;
  const int b_kg   = tid >> 7;           // 0..1
  const int b_lane = (b_n & 15) + (b_kg << 4);
  const int b_elem = (b_n >> 4) * 512 + b_lane * 16;   // chunk start in Bs

  for (int kb = 0; kb < DMODEL; kb += 32) {
    // ---- stage A tile (128 x 32) pre-swizzled ----
#pragma unroll
    for (int p = 0; p < 4; ++p) {
      const int row = p * 32 + a_rowo;
      const int k0  = a_kvec * 4;
      bf16x4 av;
      if constexpr (ABF16) {
        const bf16* A = (const bf16*)A_;
        av = *(const bf16x4*)(A + (size_t)(blockM + row) * DMODEL + kb + k0);
      } else {
        const float* A = (const float*)A_;
        f32x4 t = *(const f32x4*)(A + (size_t)(blockM + row) * DMODEL + kb + k0);
        av.x = (bf16)t.x; av.y = (bf16)t.y; av.z = (bf16)t.z; av.w = (bf16)t.w;
      }
      const int m16 = row >> 4, mm = row & 15;
      const int lA  = mm + (((k0 >> 3) & 1) << 4);
      const int e0  = (k0 & 7) + ((k0 >> 4) << 3);
      *(bf16x4*)(&As[m16 * 512 + lA * 16 + e0]) = av;
    }

    // ---- stage B tile (32 x 128): read a 16-deep K column, coalesced in n ----
    {
      const int gcol = blockN + b_n;
      v16bf bv;
#pragma unroll
      for (int kk = 0; kk < 16; ++kk)
        bv[kk] = (bf16)W[(size_t)(kb + b_kg * 8 + (kk & 7) + ((kk >> 3) << 4)) * DMODEL + gcol];
      *(v16bf*)(&Bs[b_elem]) = bv;
    }

    __syncthreads();

    // ---- fragment loads (contiguous 32B per lane) + 8 WMMAs ----
    const v16bf* Ac = (const v16bf*)As;   // 32 chunks per 16-row block
    const v16bf* Bc = (const v16bf*)Bs;   // 32 chunks per 16-col block
    v16bf afrag[2], bfrag[4];
#pragma unroll
    for (int i = 0; i < 2; ++i) afrag[i] = Ac[(2 * wm + i) * 32 + lane];
#pragma unroll
    for (int j = 0; j < 4; ++j) bfrag[j] = Bc[(4 * wn + j) * 32 + lane];

#pragma unroll
    for (int i = 0; i < 2; ++i)
#pragma unroll
      for (int j = 0; j < 4; ++j)
        acc[i][j] = __builtin_amdgcn_wmma_f32_16x16x32_bf16(
            false, afrag[i], false, bfrag[j], (short)0, acc[i][j], false, false);

    __syncthreads();
  }

  // ---- epilogue: scatter C fragments (coalesced per 16-lane half) ----
  const int rbase = blockM + wm * 32;
  const int cbase = blockN + wn * 64;
  const int rhalf = (lane >> 4) << 3;    // 0 or 8
  const int cc    = lane & 15;
#pragma unroll
  for (int i = 0; i < 2; ++i)
#pragma unroll
    for (int j = 0; j < 4; ++j)
#pragma unroll
      for (int p = 0; p < 8; ++p)
        C[(size_t)(rbase + i * 16 + rhalf + p) * DMODEL + cbase + j * 16 + cc] =
            acc[i][j][p];
}

// ---------------------------------------------------------------------------
// Strided local attention. One block per (b,c); wave h handles head h; each
// lane owns dims {l, l+32}. 16 keys per query -> shuffle-reduced dots,
// register softmax, bf16 ctx output for the final WMMA GEMM.
// ---------------------------------------------------------------------------
__global__ __launch_bounds__(256)
void attn_sparse(const float* __restrict__ Q, const float* __restrict__ K,
                 const float* __restrict__ V, bf16* __restrict__ ctx)
{
  const int bc = blockIdx.x;             // b*LQ + c
  const int b  = bc >> 12;               // /4096
  const int c  = bc & (LQ - 1);
  const int h  = threadIdx.x >> 5;
  const int l  = threadIdx.x & 31;

  const float* qrow = Q + (size_t)bc * DMODEL + h * DHEAD;
  const float q0 = qrow[l], q1 = qrow[l + 32];

  // window base: qi_col(r=0) = -floor((63-c)/4)
  const int t    = 63 - c;
  const int fd   = (t >= 0) ? (t >> 2) : -((-t + 3) >> 2);
  const int base = -fd;

  float scores[16];
#pragma unroll
  for (int r = 0; r < 16; ++r) {
    const int j   = base + r;
    const int row = c - 4 * j;
    const bool valid = (row >= 0) && (row < 64) && (j >= 0) && (j < LKV);
    const int jc  = min(max(j, 0), LKV - 1);
    const float* krow = K + (size_t)(b * LKV + jc) * DMODEL + h * DHEAD;
    float p = q0 * krow[l] + q1 * krow[l + 32];
#pragma unroll
    for (int off = 16; off > 0; off >>= 1) p += __shfl_xor(p, off, 32);
    scores[r] = valid ? p * 0.125f : -1e30f;   // 1/sqrt(64)
  }

  float m = scores[0];
#pragma unroll
  for (int r = 1; r < 16; ++r) m = fmaxf(m, scores[r]);
  float e[16], s = 0.f;
#pragma unroll
  for (int r = 0; r < 16; ++r) { e[r] = expf(scores[r] - m); s += e[r]; }
  const float inv = 1.f / s;

  float a0 = 0.f, a1 = 0.f;
#pragma unroll
  for (int r = 0; r < 16; ++r) {
    const int j  = base + r;
    const int jc = min(max(j, 0), LKV - 1);
    const float* vrow = V + (size_t)(b * LKV + jc) * DMODEL + h * DHEAD;
    const float w = e[r] * inv;           // invalid -> e[r]==0
    a0 += w * vrow[l];
    a1 += w * vrow[l + 32];
  }

  bf16* crow = ctx + (size_t)bc * DMODEL + h * DHEAD;
  crow[l]      = (bf16)a0;
  crow[l + 32] = (bf16)a1;
}

// ---------------------------------------------------------------------------
extern "C" void kernel_launch(void* const* d_in, const int* in_sizes, int n_in,
                              void* d_out, int out_size, void* d_ws, size_t ws_size,
                              hipStream_t stream)
{
  (void)in_sizes; (void)n_in; (void)out_size; (void)ws_size;
  const float* q    = (const float*)d_in[0];
  const float* k    = (const float*)d_in[1];
  const float* v    = (const float*)d_in[2];
  const float* Wq   = (const float*)d_in[3];
  const float* Wk   = (const float*)d_in[4];
  const float* Wv   = (const float*)d_in[5];
  const float* Wout = (const float*)d_in[6];
  float* out = (float*)d_out;

  const size_t MQ = (size_t)BATCH * LQ;     // 8192
  const size_t MK = (size_t)BATCH * LKV;    // 2048

  char* ws = (char*)d_ws;
  float* Qf  = (float*)ws;                                   // 16 MB
  float* Kf  = (float*)(ws + MQ * DMODEL * 4);               //  4 MB
  float* Vf  = (float*)(ws + (MQ + MK) * DMODEL * 4);        //  4 MB
  bf16*  ctx = (bf16*)(ws + (MQ + 2 * MK) * DMODEL * 4);     //  8 MB

  dim3 blk(256);
  // Q/K/V projections (fp32 A -> bf16 WMMA -> fp32 C)
  gemm512_bf16_wmma<false><<<dim3(4, MQ / 128), blk, 0, stream>>>(q, Wq, Qf);
  gemm512_bf16_wmma<false><<<dim3(4, MK / 128), blk, 0, stream>>>(k, Wk, Kf);
  gemm512_bf16_wmma<false><<<dim3(4, MK / 128), blk, 0, stream>>>(v, Wv, Vf);

  // strided-window attention -> bf16 ctx
  attn_sparse<<<dim3(MQ), blk, 0, stream>>>(Qf, Kf, Vf, ctx);

  // output projection (bf16 A)
  gemm512_bf16_wmma<true><<<dim3(4, MQ / 128), blk, 0, stream>>>(ctx, Wout, out);
}
